// EnergyLinearCQN_68135361184375
// MI455X (gfx1250) — matrix operand order repaired and checked by reading, not compiled, not run on hardware
//
#include <hip/hip_runtime.h>

// ---------------- problem constants (from reference) ----------------
#define B_ROWS  65536
#define D_IN    32
#define H1N     512
#define H2N     1024
#define OUTN    64
#define S_NUM   8192
#define INV_ALPHA 10.0f   // 1/0.1

typedef __attribute__((ext_vector_type(16))) __bf16 v16bf;
typedef __attribute__((ext_vector_type(8)))  float  v8f;

union AFrag { v16bf v; uint4 u[2]; };

__device__ __forceinline__ unsigned short f2bf(float f) {
    unsigned int x = __float_as_uint(f);
    unsigned int r = x + 0x7fffu + ((x >> 16) & 1u);   // round-to-nearest-even
    return (unsigned short)(r >> 16);
}

// ---------------- pack X = concat(state, preference) -> bf16 [B,32] ----------------
__global__ __launch_bounds__(256)
void pack_x_kernel(const float* __restrict__ state, const float* __restrict__ pref,
                   unsigned short* __restrict__ xout) {
    size_t idx = (size_t)blockIdx.x * 256 + threadIdx.x;   // B*32 threads
    int row = (int)(idx >> 5);
    int col = (int)(idx & 31);
    float v = (col < 30) ? state[(size_t)row * 30 + col]
                         : pref[(size_t)row * 2 + (col - 30)];
    xout[idx] = f2bf(v);
}

// ---------------- pack W [K,N] f32 row-major -> WMMA B-fragment tiles (bf16) ----------
// Tile (kt,nt): 32(K) x 16(N). Tile order: nt-major (t = nt*(K/32) + kt).
// Within tile: dword d = lane*8 + j holds B[k0,n],B[k0+1,n] with
//   half = lane>>4, col = lane&15, k0 = kt*32 + 16*half + 2*j, n = nt*16 + col.
__global__ __launch_bounds__(256)
void pack_w_kernel(const float* __restrict__ W, unsigned short* __restrict__ Wp,
                   int K, int N) {
    size_t idx = (size_t)blockIdx.x * 256 + threadIdx.x;   // one dword each
    size_t total = ((size_t)K * N) >> 1;
    if (idx >= total) return;
    int tile   = (int)(idx >> 8);
    int within = (int)(idx & 255);
    int lane = within >> 3;
    int j    = within & 7;
    int ktiles = K >> 5;
    int nt = tile / ktiles;
    int kt = tile - nt * ktiles;
    int half = lane >> 4;
    int col  = lane & 15;
    int k0 = kt * 32 + 16 * half + 2 * j;
    int n  = nt * 16 + col;
    Wp[idx * 2 + 0] = f2bf(W[(size_t)k0 * N + n]);
    Wp[idx * 2 + 1] = f2bf(W[(size_t)(k0 + 1) * N + n]);
}

// ---------------- fused GEMM + bias (+ReLU): out = act(A[M,K] * W[K,N] + b) -------
// Block: 256 threads = 8 waves; wave w owns rows [blockIdx.x*128 + w*16, +16),
// cols [blockIdx.y*64, +64) as 4 accumulators.
// Weight fragments for the block's 4 N-tiles are shared by all 8 waves, so they are
// staged in LDS via async DMA (global_load_async_to_lds_b128, ASYNCcnt), double-
// buffered in chunks of CKT k-tiles (CKT*4 KB per buffer) and overlapped with WMMA.
template<int CKT>
__global__ __launch_bounds__(256)
void mlp_gemm_kernel(const unsigned short* __restrict__ A,   // bf16 [M,K] row-major
                     const unsigned short* __restrict__ Wp,  // packed fragments
                     const float* __restrict__ bias,         // [N]
                     unsigned short* __restrict__ outb,      // bf16 [M,N] or null
                     float* __restrict__ outf,               // f32  [M,N] or null
                     int M, int K, int N, int relu) {
    __shared__ uint4 smem[2 * CKT * 256];   // double-buffered B chunk (4 tiles x CKT kt)

    const int tid  = threadIdx.x;
    const int lane = tid & 31;
    const int wv   = tid >> 5;
    const int m0   = blockIdx.x * 128 + wv * 16;
    const int half = lane >> 4;
    const int row  = m0 + (lane & 15);
    const int ktiles  = K >> 5;
    const int nchunks = ktiles / CKT;
    const int nt0  = blockIdx.y * 4;

    const unsigned short* arow = A + (size_t)row * K;
    // low 32 bits of a flat LDS pointer == wave-relative LDS byte offset
    const unsigned lds_base = (unsigned)(size_t)(&smem[0]);

    // cooperative async-DMA of B chunk `kc` into buffer `buf`
    auto issue_chunk = [&](int kc, int buf) {
        const int k0 = kc * CKT;
#pragma unroll
        for (int i = 0; i < CKT; ++i) {
            int u   = i * 256 + tid;          // 16-byte unit within chunk
            int seg = u / (CKT * 64);         // which of the 4 N-tiles
            int rem = u - seg * (CKT * 64);   // unit within that tile's CKT KB
            const unsigned short* g =
                Wp + ((size_t)(nt0 + seg) * ktiles + k0) * 512 + (size_t)rem * 8;
            unsigned ldsoff = lds_base + (unsigned)(buf * (CKT * 4096) + u * 16);
            asm volatile("global_load_async_to_lds_b128 %0, %1, off"
                         :: "v"(ldsoff), "v"(g) : "memory");
        }
    };

    v8f acc[4] = {};

    issue_chunk(0, 0);
    asm volatile("s_wait_asynccnt 0" ::: "memory");
    __syncthreads();

    for (int kc = 0; kc < nchunks; ++kc) {
        const int buf = kc & 1;
        if (kc + 1 < nchunks) issue_chunk(kc + 1, buf ^ 1);   // overlap DMA w/ compute

        const int k0 = kc * CKT;
        const uint4* bb = smem + buf * (CKT * 256);
#pragma unroll
        for (int c = 0; c < CKT; ++c) {
            // A fragment: lane holds row (lane&15); K runs {16h..16h+15}
            AFrag a;
            const unsigned short* ap = arow + (k0 + c) * 32 + 16 * half;
            a.u[0] = *(const uint4*)(ap);
            a.u[1] = *(const uint4*)(ap + 8);
#pragma unroll
            for (int t = 0; t < 4; ++t) {
                AFrag b;
                int bi = (t * CKT + c) * 64 + lane * 2;
                b.u[0] = bb[bi];
                b.u[1] = bb[bi + 1];
                acc[t] = __builtin_amdgcn_wmma_f32_16x16x32_bf16(
                    false, a.v, false, b.v, (short)0, acc[t], false, false);
            }
        }

        asm volatile("s_wait_asynccnt 0" ::: "memory");
        __syncthreads();
    }

    // Epilogue: C/D layout -> VGPR r: row = m0 + r + 8*half, col = lane&15
    const int col = lane & 15;
#pragma unroll
    for (int t = 0; t < 4; ++t) {
        int n = (nt0 + t) * 16 + col;
        float bv = bias[n];
#pragma unroll
        for (int r = 0; r < 8; ++r) {
            float v = acc[t][r] + bv;
            if (relu) v = fmaxf(v, 0.0f);
            size_t o = (size_t)(m0 + r + 8 * half) * N + n;
            if (outb) outb[o] = f2bf(v);
            else      outf[o] = v;
        }
    }
}

// ---------------- envelope H(): softmax combine over 256 (action,w) candidates ----
// Rows b = s + wv*8192 (wv=0..7) share the same candidate set -> stage in LDS once.
__global__ __launch_bounds__(256)
void envelope_kernel(const float* __restrict__ q,     // [B, 64] f32
                     const float* __restrict__ pref,  // [B, 2]
                     float* __restrict__ hq) {        // [B, 2]
    __shared__ float2 qs[256];
    const int s = blockIdx.x;
    const int t = threadIdx.x;
    {   // candidate j = w*32 + a comes from q[s + w*8192, a, :]
        int w = t >> 5, a = t & 31;
        const float* qr = q + ((size_t)(s + w * S_NUM)) * 64 + a * 2;
        qs[t] = make_float2(qr[0], qr[1]);
    }
    __syncthreads();

    const int lane = t & 31;
    const int wv   = t >> 5;
    const int b    = s + wv * S_NUM;
    const float px = pref[(size_t)b * 2 + 0];
    const float py = pref[(size_t)b * 2 + 1];

    float dots[8];
    float2 qv[8];
    float m = -3.4e38f;
#pragma unroll
    for (int i = 0; i < 8; ++i) {
        float2 c = qs[i * 32 + lane];
        qv[i] = c;
        dots[i] = c.x * px + c.y * py;
        m = fmaxf(m, dots[i]);
    }
#pragma unroll
    for (int off = 16; off > 0; off >>= 1)
        m = fmaxf(m, __shfl_xor(m, off, 32));

    float se = 0.f, sx = 0.f, sy = 0.f;
#pragma unroll
    for (int i = 0; i < 8; ++i) {
        float e = __expf((dots[i] - m) * INV_ALPHA);
        se += e; sx += e * qv[i].x; sy += e * qv[i].y;
    }
#pragma unroll
    for (int off = 16; off > 0; off >>= 1) {
        se += __shfl_xor(se, off, 32);
        sx += __shfl_xor(sx, off, 32);
        sy += __shfl_xor(sy, off, 32);
    }
    if (lane == 0) {
        hq[(size_t)b * 2 + 0] = sx / se;
        hq[(size_t)b * 2 + 1] = sy / se;
    }
}

// ---------------------------------------------------------------------------------
extern "C" void kernel_launch(void* const* d_in, const int* in_sizes, int n_in,
                              void* d_out, int out_size, void* d_ws, size_t ws_size,
                              hipStream_t stream) {
    const float* state = (const float*)d_in[0];
    const float* pref  = (const float*)d_in[1];
    const float* W1 = (const float*)d_in[2];
    const float* b1 = (const float*)d_in[3];
    const float* W2 = (const float*)d_in[4];
    const float* b2 = (const float*)d_in[5];
    const float* W3 = (const float*)d_in[6];
    const float* b3 = (const float*)d_in[7];
    (void)in_sizes; (void)n_in; (void)out_size; (void)ws_size;

    // workspace carve-up (bf16 buffers stored as ushort)
    char* ws = (char*)d_ws;
    size_t off = 0;
    auto take = [&](size_t bytes) { char* p = ws + off; off = (off + bytes + 255) & ~(size_t)255; return p; };
    unsigned short* Xin = (unsigned short*)take((size_t)B_ROWS * D_IN * 2);
    unsigned short* X1  = (unsigned short*)take((size_t)B_ROWS * H1N  * 2);
    unsigned short* X2  = (unsigned short*)take((size_t)B_ROWS * H2N  * 2);
    unsigned short* W1p = (unsigned short*)take((size_t)D_IN * H1N * 2);
    unsigned short* W2p = (unsigned short*)take((size_t)H1N  * H2N * 2);
    unsigned short* W3p = (unsigned short*)take((size_t)H2N  * OUTN * 2);

    float* out  = (float*)d_out;
    float* hq   = out;                         // [B, 2]
    float* qout = out + (size_t)B_ROWS * 2;    // [B, 64] == q.reshape(B, 32, 2)

    // 1) pack input and weights to bf16 (fragment-major for weights)
    pack_x_kernel<<<(B_ROWS * D_IN) / 256, 256, 0, stream>>>(state, pref, Xin);
    pack_w_kernel<<<((size_t)D_IN * H1N / 2 + 255) / 256, 256, 0, stream>>>(W1, W1p, D_IN, H1N);
    pack_w_kernel<<<((size_t)H1N  * H2N / 2 + 255) / 256, 256, 0, stream>>>(W2, W2p, H1N, H2N);
    pack_w_kernel<<<((size_t)H2N * OUTN / 2 + 255) / 256, 256, 0, stream>>>(W3, W3p, H2N, OUTN);

    // 2) three WMMA GEMM layers (async-LDS staged weights)
    {   dim3 g(B_ROWS / 128, H1N / 64);
        mlp_gemm_kernel<1><<<g, 256, 0, stream>>>(Xin, W1p, b1, X1, nullptr,
                                                  B_ROWS, D_IN, H1N, 1); }
    {   dim3 g(B_ROWS / 128, H2N / 64);
        mlp_gemm_kernel<4><<<g, 256, 0, stream>>>(X1, W2p, b2, X2, nullptr,
                                                  B_ROWS, H1N, H2N, 1); }
    {   dim3 g(B_ROWS / 128, OUTN / 64);
        mlp_gemm_kernel<4><<<g, 256, 0, stream>>>(X2, W3p, b3, nullptr, qout,
                                                  B_ROWS, H2N, OUTN, 0); }

    // 3) envelope combine -> hq
    envelope_kernel<<<S_NUM, 256, 0, stream>>>(qout, pref, hq);
}
